// GroupedAtomMoE_53618371723357
// MI455X (gfx1250) — compile-verified, matrix operand order checked by reading, not run on hardware
//
#include <hip/hip_runtime.h>
#include <math.h>
#include <stdint.h>

// ---------------- constants (from reference) ----------------
#define TKN   4096          // B*N = 8*512 tokens
#define DF    512
#define HGATE 256
#define NEXP  4
#define DH    2048
#define TEMPV 1.2f

typedef __attribute__((ext_vector_type(16))) __bf16 bfvec16;
typedef __attribute__((ext_vector_type(8)))  float  v8f;

__device__ __forceinline__ float gelu_erf(float x) {
    return 0.5f * x * (1.0f + erff(x * 0.70710678118654752f));
}

// ---------------- bf16 WMMA GEMM --------------------------------------------
// C[M,N] = op( A[M,Kd](bf16,row-major) * BT[N,Kd](bf16,row-major==B col-major) + bias )
// flags bit0: GELU epilogue
// flags bit1: accumulate  outF[r,c] += rowscale[r*rs_stride] * val   (rowscale!=null)
//
// Block: 128 threads = 4 waves. Block tile 128x64; wave w owns rows [m0+32w, +32).
// B panel staged in LDS (fragment layout, 2 stages); K-loop manually unrolled by 2
// with disjoint ping-pong A register sets (aA/aB) written directly by loads.
// All 4 B fragments of a substep are loaded into distinct register sets before the
// WMMA burst so ds waits can be partial (in-order LDS returns). Requires Kd % 64 == 0.
__global__ __launch_bounds__(128)
void gemm_bf16_wmma(const __bf16* __restrict__ A, const __bf16* __restrict__ BT,
                    const float* __restrict__ bias,
                    const float* __restrict__ rowscale, int rs_stride,
                    float* __restrict__ outF, __bf16* __restrict__ outB,
                    int M, int N, int Kd, int flags) {
    __shared__ uint4 ldsB[512];              // 2 stages x (4 frags x 32 lanes x 2 uint4) = 8 KB
    const int tid  = threadIdx.x;
    const int lane = tid & 31;
    const int wave = tid >> 5;
    const int r    = lane & 15;              // row (A) / col (B) within 16-tile
    const int half = lane >> 4;              // K-half selector per ISA 16-bit layout
    const int m0   = blockIdx.x * 128 + wave * 32;
    const int n0   = blockIdx.y * 64;

    // cooperative B loader: thread t -> (frag nt = t>>5, frag-lane bl = t&31)
    const unsigned short* Bsrc = (const unsigned short*)BT
        + (size_t)(n0 + (tid >> 5) * 16 + (tid & 15)) * Kd + (((tid >> 4) & 1) * 8);

    const unsigned short* A0 = (const unsigned short*)A + (size_t)(m0 + r) * Kd + half * 8;
    const unsigned short* A1 = A0 + (size_t)16 * Kd;

    union Frag { bfvec16 v; uint4 q[2]; };
    v8f acc[8] = {};

    auto loadA = [&](int k, uint4* dst) {
        dst[0] = *(const uint4*)(A0 + k);
        dst[1] = *(const uint4*)(A0 + k + 16);
        dst[2] = *(const uint4*)(A1 + k);
        dst[3] = *(const uint4*)(A1 + k + 16);
    };
    auto loadB = [&](int k, int stage) {
        ldsB[stage * 256 + tid * 2]     = *(const uint4*)(Bsrc + k);
        ldsB[stage * 256 + tid * 2 + 1] = *(const uint4*)(Bsrc + k + 16);
    };
    auto compute = [&](const uint4* av, int stage) {
        Frag a0, a1;
        a0.q[0] = av[0]; a0.q[1] = av[1];
        a1.q[0] = av[2]; a1.q[1] = av[3];
        const uint4* bs = &ldsB[stage * 256 + lane * 2];
        // four distinct B fragment register sets, loaded before the WMMA burst;
        // LDS returns in order -> scheduler can use partial s_wait_dscnt
        Frag b0, b1, b2, b3;
        b0.q[0] = bs[0];   b0.q[1] = bs[1];
        b1.q[0] = bs[64];  b1.q[1] = bs[65];
        b2.q[0] = bs[128]; b2.q[1] = bs[129];
        b3.q[0] = bs[192]; b3.q[1] = bs[193];
        acc[0] = __builtin_amdgcn_wmma_f32_16x16x32_bf16(false, a0.v, false, b0.v, (short)0, acc[0], false, false);
        acc[4] = __builtin_amdgcn_wmma_f32_16x16x32_bf16(false, a1.v, false, b0.v, (short)0, acc[4], false, false);
        acc[1] = __builtin_amdgcn_wmma_f32_16x16x32_bf16(false, a0.v, false, b1.v, (short)0, acc[1], false, false);
        acc[5] = __builtin_amdgcn_wmma_f32_16x16x32_bf16(false, a1.v, false, b1.v, (short)0, acc[5], false, false);
        acc[2] = __builtin_amdgcn_wmma_f32_16x16x32_bf16(false, a0.v, false, b2.v, (short)0, acc[2], false, false);
        acc[6] = __builtin_amdgcn_wmma_f32_16x16x32_bf16(false, a1.v, false, b2.v, (short)0, acc[6], false, false);
        acc[3] = __builtin_amdgcn_wmma_f32_16x16x32_bf16(false, a0.v, false, b3.v, (short)0, acc[3], false, false);
        acc[7] = __builtin_amdgcn_wmma_f32_16x16x32_bf16(false, a1.v, false, b3.v, (short)0, acc[7], false, false);
    };

    // ---- prologue
    uint4 aA[4], aB[4];
    loadA(0, aA);
    loadB(0, 0);
    __syncthreads();

    // ---- main loop, 2 K-steps (64) per iteration, ping-pong regs + LDS stages
    for (int k = 0; k < Kd; k += 64) {
        loadA(k + 32, aB);                    // always valid (k+32 <= Kd-32)
        loadB(k + 32, 1);
        __builtin_prefetch(A0 + k + 64, 0, 1);            // global_prefetch_b8
        compute(aA, 0);
        __syncthreads();                      // stage1 written + stage0 consumed

        if (k + 64 < Kd) {
            loadA(k + 64, aA);
            loadB(k + 64, 0);
            __builtin_prefetch(A1 + k + 96, 0, 1);
        }
        compute(aB, 1);
        __syncthreads();                      // stage0 written + stage1 consumed
    }

    // ---- epilogue: VGPR i -> row m0 + hm*16 + half*8 + i ; lane%16 -> col
#pragma unroll
    for (int hm = 0; hm < 2; ++hm) {
#pragma unroll
        for (int nt = 0; nt < 4; ++nt) {
            const v8f c = acc[hm * 4 + nt];
            const int col = n0 + nt * 16 + r;
            const float bv = bias ? bias[col] : 0.0f;
#pragma unroll
            for (int i = 0; i < 8; ++i) {
                const int row = m0 + hm * 16 + half * 8 + i;
                float val = c[i] + bv;
                if (flags & 1) val = gelu_erf(val);
                const size_t o = (size_t)row * N + col;
                if (flags & 2) {
                    const float s = rowscale[(size_t)row * rs_stride];
                    outF[o] += s * val;
                } else {
                    if (outF) outF[o] = val;
                    if (outB) outB[o] = (__bf16)val;
                }
            }
        }
    }
}

// ---------------- elementwise helpers ---------------------------------------
__global__ void cvt_f32_bf16_k(const float* __restrict__ s, __bf16* __restrict__ d, int n) {
    int i = blockIdx.x * 256 + threadIdx.x;
    if (i < n) d[i] = (__bf16)s[i];
}
__global__ void copy_f32_k(const float* __restrict__ s, float* __restrict__ d, int n) {
    int i = blockIdx.x * 256 + threadIdx.x;
    if (i < n) d[i] = s[i];
}
// dst[C,R] (bf16) = transpose(src[R,C] f32)
__global__ void transpose_cvt_k(const float* __restrict__ s, __bf16* __restrict__ d, int R, int C) {
    int i = blockIdx.x * 256 + threadIdx.x;
    if (i >= R * C) return;
    int c = i / R, r = i % R;
    d[(size_t)c * R + r] = (__bf16)s[(size_t)r * C + c];
}
// xcat = concat(a[T,Da], b[T,Db]) -> bf16 [T, Da+Db]
__global__ void concat_cvt_k(const float* __restrict__ a, const float* __restrict__ b,
                             __bf16* __restrict__ d, int T, int Da, int Db) {
    int D = Da + Db;
    int i = blockIdx.x * 256 + threadIdx.x;
    if (i >= T * D) return;
    int t = i / D, j = i % D;
    float v = (j < Da) ? a[(size_t)t * Da + j] : b[(size_t)t * Db + (j - Da)];
    d[i] = (__bf16)v;
}

// ---------------- LayerNorm (+optional GELU) -> bf16 -------------------------
__global__ __launch_bounds__(256)
void ln_k(const float* __restrict__ x, const float* __restrict__ g,
          const float* __restrict__ b, __bf16* __restrict__ out, int D, int do_gelu) {
    const int row = blockIdx.x;
    const float* xr = x + (size_t)row * D;
    __shared__ float s1[8], s2[8];
    float ls = 0.f, lq = 0.f;
    for (int i = threadIdx.x; i < D; i += 256) { float v = xr[i]; ls += v; lq += v * v; }
    for (int off = 16; off; off >>= 1) {
        ls += __shfl_down(ls, off, 32);
        lq += __shfl_down(lq, off, 32);
    }
    const int wv = threadIdx.x >> 5, ln = threadIdx.x & 31;
    if (ln == 0) { s1[wv] = ls; s2[wv] = lq; }
    __syncthreads();
    if (threadIdx.x == 0) {
        float a = 0.f, c = 0.f;
        for (int i = 0; i < 8; ++i) { a += s1[i]; c += s2[i]; }
        s1[0] = a; s2[0] = c;
    }
    __syncthreads();
    const float mean = s1[0] / D;
    const float var  = s2[0] / D - mean * mean;
    const float rstd = rsqrtf(var + 1e-5f);
    __bf16* orow = out + (size_t)row * D;
    for (int i = threadIdx.x; i < D; i += 256) {
        float v = (xr[i] - mean) * rstd * g[i] + b[i];
        if (do_gelu) v = gelu_erf(v);
        orow[i] = (__bf16)v;
    }
}

// ---------------- gate heads --------------------------------------------------
// gl[t] = (dot(g1[t,:H], Wg2) + bg2)/TEMP
__global__ void gl_k(const __bf16* __restrict__ g1, const float* __restrict__ Wg2,
                     const float* __restrict__ bg2, float* __restrict__ gl, int T, int H) {
    int t = blockIdx.x * 256 + threadIdx.x;
    if (t >= T) return;
    const __bf16* row = g1 + (size_t)t * H;
    float s = 0.f;
    for (int i = 0; i < H; ++i) s += (float)row[i] * Wg2[i];
    gl[t] = (s + bg2[0]) / TEMPV;
}
// el = e1 @ We2 + be2; mask; /TEMP; top-2 softmax -> probs[T,NEXP]
__global__ void probs_k(const __bf16* __restrict__ e1, const float* __restrict__ We2,
                        const float* __restrict__ be2, const unsigned char* __restrict__ mask,
                        float* __restrict__ probs, int T, int H) {
    int t = blockIdx.x * 256 + threadIdx.x;
    if (t >= T) return;
    float el[NEXP];
    for (int k = 0; k < NEXP; ++k) el[k] = be2[k];
    const __bf16* row = e1 + (size_t)t * H;
    for (int i = 0; i < H; ++i) {
        float v = (float)row[i];
        for (int k = 0; k < NEXP; ++k) el[k] += v * We2[i * NEXP + k];
    }
    float* p = probs + (size_t)t * NEXP;
    if (!mask[t]) { for (int k = 0; k < NEXP; ++k) p[k] = 0.f; return; }
    for (int k = 0; k < NEXP; ++k) el[k] /= TEMPV;
    int i1 = 0;
    for (int k = 1; k < NEXP; ++k) if (el[k] > el[i1]) i1 = k;     // first max on ties
    int i2 = -1;
    for (int k = 0; k < NEXP; ++k) if (k != i1 && (i2 < 0 || el[k] > el[i2])) i2 = k;
    float w2 = __expf(el[i2] - el[i1]);
    float s  = 1.0f + w2;
    for (int k = 0; k < NEXP; ++k) p[k] = 0.f;
    p[i1] = 1.0f / s; p[i2] = w2 / s;
}

// ---------------- final group-softmax blend ----------------------------------
__global__ void combine_k(const float* __restrict__ o0, const float* __restrict__ o1,
                          const float* __restrict__ o2, const float* __restrict__ gl,
                          const unsigned char* __restrict__ mask,
                          float* __restrict__ out, int T, int D) {
    int i = blockIdx.x * 256 + threadIdx.x;
    if (i >= T * D) return;
    int t = i / D;
    if (!mask[t]) { out[i] = 0.f; return; }
    float l0 = gl[t], l1 = gl[T + t], l2 = gl[2 * T + t];
    float mx = fmaxf(l0, fmaxf(l1, l2));
    float e0 = __expf(l0 - mx), e1 = __expf(l1 - mx), e2 = __expf(l2 - mx);
    float s = e0 + e1 + e2;
    out[i] = (o0[i] * e0 + o1[i] * e1 + o2[i] * e2) / s;
}

// ---------------- host side ---------------------------------------------------
static inline void launch_gemm(const __bf16* A, const __bf16* BT, const float* bias,
                               const float* rs, int rss, float* oF, __bf16* oB,
                               int M, int N, int Kd, int flags, hipStream_t st) {
    dim3 grid(M / 128, N / 64);
    gemm_bf16_wmma<<<grid, 128, 0, st>>>(A, BT, bias, rs, rss, oF, oB, M, N, Kd, flags);
}
static inline int cdiv(int a, int b) { return (a + b - 1) / b; }

extern "C" void kernel_launch(void* const* d_in, const int* in_sizes, int n_in,
                              void* d_out, int out_size, void* d_ws, size_t ws_size,
                              hipStream_t stream) {
    (void)in_sizes; (void)n_in; (void)out_size; (void)ws_size;
    const int T = TKN;

    // ---- input map (setup_inputs dict order, group dicts flattened in order)
    const float* h2d = (const float*)d_in[0];
    const float* h3d = (const float*)d_in[1];
    const unsigned char* mask = (const unsigned char*)d_in[2];
    // per-group leaf order:
    // 0 Wp 1 bp 2 ln_s_g 3 ln_s_b 4 Ws 5 bs 6 ln_g_g 7 ln_g_b 8 Wg1 9 bg1 10 Wg2 11 bg2
    // 12 ln_e_g 13 ln_e_b 14 We1 15 be1 16 We2 17 be2 18 EW1 19 Eb1 20 EW2 21 Eb2
    const float* w_out = (const float*)d_in[3 + 3 * 22 + 0];
    const float* b_out = (const float*)d_in[3 + 3 * 22 + 1];

    // ---- workspace bump allocator
    uintptr_t cur = (uintptr_t)d_ws;
    auto alloc = [&](size_t bytes) -> void* {
        uintptr_t r = (cur + 255) & ~(uintptr_t)255;
        cur = r + bytes;
        return (void*)r;
    };
    __bf16* x2b   = (__bf16*)alloc((size_t)T * 256 * 2);
    __bf16* x3b   = (__bf16*)alloc((size_t)T * 256 * 2);
    __bf16* xcb   = (__bf16*)alloc((size_t)T * 512 * 2);
    __bf16* WpT   = (__bf16*)alloc((size_t)DF * 512 * 2);        // max d_in = 512
    __bf16* WsT   = (__bf16*)alloc((size_t)DF * DF * 2);
    __bf16* Wg1T  = (__bf16*)alloc((size_t)HGATE * DF * 2);
    __bf16* We1T  = (__bf16*)alloc((size_t)HGATE * DF * 2);
    __bf16* EW1T  = (__bf16*)alloc((size_t)NEXP * DH * DF * 2);
    __bf16* EW2T  = (__bf16*)alloc((size_t)NEXP * DF * DH * 2);
    __bf16* WoT   = (__bf16*)alloc((size_t)DF * DF * 2);
    float*  tmpf  = (float*) alloc((size_t)T * DF * 4);
    __bf16* sb    = (__bf16*)alloc((size_t)T * DF * 2);          // shared bf16 scratch
    float*  basef = (float*) alloc((size_t)T * DF * 4);
    __bf16* baseb = (__bf16*)alloc((size_t)T * DF * 2);
    __bf16* g1e1  = (__bf16*)alloc((size_t)T * HGATE * 2);
    __bf16* h1    = (__bf16*)alloc((size_t)T * DH * 2);
    float*  accf  = (float*) alloc((size_t)T * DF * 4);
    float*  probs = (float*) alloc((size_t)T * NEXP * 4);
    float*  glb   = (float*) alloc((size_t)3 * T * 4);
    float*  ob    = (float*) alloc((size_t)3 * T * DF * 4);

    // ---- one-time per-call conversions
    cvt_f32_bf16_k<<<cdiv(T * 256, 256), 256, 0, stream>>>(h2d, x2b, T * 256);
    cvt_f32_bf16_k<<<cdiv(T * 256, 256), 256, 0, stream>>>(h3d, x3b, T * 256);
    concat_cvt_k<<<cdiv(T * 512, 256), 256, 0, stream>>>(h2d, h3d, xcb, T, 256, 256);
    transpose_cvt_k<<<cdiv(DF * DF, 256), 256, 0, stream>>>(w_out, WoT, DF, DF);

    for (int g = 0; g < 3; ++g) {
        const int gb  = 3 + 22 * g;
        const int din = (g == 2) ? 512 : 256;
        const __bf16* X = (g == 0) ? x2b : (g == 1) ? x3b : xcb;
        auto P = [&](int i) { return (const float*)d_in[gb + i]; };
        const float *Wp = P(0), *bp = P(1), *lnsg = P(2), *lnsb = P(3), *Ws = P(4), *bs = P(5);
        const float *lngg = P(6), *lngb = P(7), *Wg1 = P(8), *bg1 = P(9), *Wg2 = P(10), *bg2 = P(11);
        const float *lneg = P(12), *lneb = P(13), *We1 = P(14), *be1 = P(15), *We2 = P(16), *be2 = P(17);
        const float *EW1 = P(18), *Eb1 = P(19), *EW2 = P(20), *Eb2 = P(21);

        // weight transposes -> bf16 (B stored as [N,K] so fragments load contiguously)
        transpose_cvt_k<<<cdiv(din * DF, 256), 256, 0, stream>>>(Wp, WpT, din, DF);
        transpose_cvt_k<<<cdiv(DF * DF, 256), 256, 0, stream>>>(Ws, WsT, DF, DF);
        transpose_cvt_k<<<cdiv(DF * HGATE, 256), 256, 0, stream>>>(Wg1, Wg1T, DF, HGATE);
        transpose_cvt_k<<<cdiv(DF * HGATE, 256), 256, 0, stream>>>(We1, We1T, DF, HGATE);
        for (int k = 0; k < NEXP; ++k) {
            transpose_cvt_k<<<cdiv(DF * DH, 256), 256, 0, stream>>>(
                EW1 + (size_t)k * DF * DH, EW1T + (size_t)k * DH * DF, DF, DH);
            transpose_cvt_k<<<cdiv(DH * DF, 256), 256, 0, stream>>>(
                EW2 + (size_t)k * DH * DF, EW2T + (size_t)k * DF * DH, DH, DF);
        }

        // shared trunk: base = GELU(LN(X@Wp+bp)) @ Ws + bs
        launch_gemm(X, WpT, bp, nullptr, 0, tmpf, nullptr, T, DF, din, 0, stream);
        ln_k<<<T, 256, 0, stream>>>(tmpf, lnsg, lnsb, sb, DF, 1);
        launch_gemm(sb, WsT, bs, nullptr, 0, basef, baseb, T, DF, DF, 0, stream);

        // group gate logit
        ln_k<<<T, 256, 0, stream>>>(basef, lngg, lngb, sb, DF, 0);
        launch_gemm(sb, Wg1T, bg1, nullptr, 0, nullptr, g1e1, T, HGATE, DF, 1, stream);
        gl_k<<<cdiv(T, 256), 256, 0, stream>>>(g1e1, Wg2, bg2, glb + (size_t)g * T, T, HGATE);

        // expert gate -> top-2 probs
        ln_k<<<T, 256, 0, stream>>>(basef, lneg, lneb, sb, DF, 0);
        launch_gemm(sb, We1T, be1, nullptr, 0, nullptr, g1e1, T, HGATE, DF, 1, stream);
        probs_k<<<cdiv(T, 256), 256, 0, stream>>>(g1e1, We2, be2, mask, probs, T, HGATE);

        // dense MoE with prob-weighted accumulate; acc starts at base (residual)
        copy_f32_k<<<cdiv(T * DF, 256), 256, 0, stream>>>(basef, accf, T * DF);
        for (int k = 0; k < NEXP; ++k) {
            launch_gemm(baseb, EW1T + (size_t)k * DH * DF, Eb1 + (size_t)k * DH,
                        nullptr, 0, nullptr, h1, T, DH, DF, 1, stream);          // GELU
            launch_gemm(h1, EW2T + (size_t)k * DF * DH, Eb2 + (size_t)k * DF,
                        probs + k, NEXP, accf, nullptr, T, DF, DH, 2, stream);   // += p*(..)
        }

        // o_g = (y + base) @ w_out + b_out
        cvt_f32_bf16_k<<<cdiv(T * DF, 256), 256, 0, stream>>>(accf, sb, T * DF);
        launch_gemm(sb, WoT, b_out, nullptr, 0, ob + (size_t)g * T * DF, nullptr,
                    T, DF, DF, 0, stream);
    }

    combine_k<<<cdiv(T * DF, 256), 256, 0, stream>>>(
        ob, ob + (size_t)T * DF, ob + (size_t)2 * T * DF, glb, mask, (float*)d_out, T, DF);
}
